// MoELayer_12455405158655
// MI455X (gfx1250) — matrix operand order repaired and checked by reading, not compiled, not run on hardware
//
#include <hip/hip_runtime.h>

// ---------------------------------------------------------------------------
// MoE forward for MI455X (gfx1250, wave32, WMMA).
// HBM-bound (~1.1 GB fp32 weights vs ~1.65 TFLOP -> ~47 us at 23.3 TB/s):
// all GEMMs use v_wmma_f32_16x16x32_bf16 with on-the-fly fp32->bf16 convert.
// A tiles are staged into LDS in *fragment order* (one v16bf per lane,
// double-buffered, 1 barrier/K-step); B fragments are built directly from
// global (no inter-wave B reuse at M-tile=16, so LDS round-trip is waste).
// bf16 intermediates live in d_ws (~135.5 MB, regions reused sequentially).
// ---------------------------------------------------------------------------

typedef __bf16 bf16_t;
typedef __bf16 v16bf __attribute__((ext_vector_type(16)));
typedef __bf16 v4bf  __attribute__((ext_vector_type(4)));
typedef float  v8f   __attribute__((ext_vector_type(8)));
typedef float  v4f   __attribute__((ext_vector_type(4)));

#define T_TOK      8192          // B*S tokens
#define HDIM       2048
#define NEXP       8
#define INTER_D    4096
#define SHARED_ID  8192
#define PAD_SLOTS  16512         // T*TOPK + NEXP*16 padding headroom
#define MAX_MTILES (PAD_SLOTS / 16)   // 1032

// ws byte layout
// meta ints: [0..7]=counts, [8..16]=padded exclusive slotBase (+total), [17..24]=cursor
#define WS_META    0
#define WS_ROWMAP  128
#define WS_TOPIDX  (WS_ROWMAP + PAD_SLOTS * 4)
#define WS_SCORES  (WS_TOPIDX + T_TOK * 2 * 4)
#define WS_SLOTSC  (WS_SCORES + T_TOK * 2 * 4)
#define WS_BIG     (((WS_SLOTSC + PAD_SLOTS * 4) + 255) & ~255)
// big bf16 buffer: first shared-expert intermediate [T_TOK x SHARED_ID],
// then reused as routed activations [PAD_SLOTS x INTER_D]. ~135.3 MB.

// ---------------------------------------------------------------------------
// WMMA fragment layouts (CDNA5 ISA 7.12.2, wave32):
// A (16x32 bf16): lane<16 -> row=lane,    K in {0..7,16..23}
//                 lane>=16 -> row=lane-16, K in {8..15,24..31}
// B as loaded here (A staged in same k order): lane<16 -> col=lane,    K=0..15 ; lane>=16 -> col, K=16..31
// C (16x16 f32):  vgpr i -> row = i + (lane<16 ? 0 : 8), col = lane&15
//
// A is staged in LDS already in fragment order: lAf[lane] is the lane's v16bf.
// Staging thread t<128 handles tile row r=t>>3, K-group K0=(t&7)*4:
//   lane = r + ((K0&16)?16:0) ; v4bf slot = (K0&15)/4
// ---------------------------------------------------------------------------

__device__ inline v16bf load_b_direct_f32(const float* __restrict__ B, int ldb,
                                          int k0, int col, int lane) {
  const float* p = B + (size_t)(k0 + (lane & 16)) * ldb + col + (lane & 15);
  v16bf b;
#pragma unroll
  for (int i = 0; i < 16; ++i) b[i] = (bf16_t)p[(size_t)i * ldb];
  return b;
}

__device__ inline float siluf(float x) { return x / (1.0f + __expf(-x)); }

// ---------------------------------------------------------------------------
// 0) init: clear meta, poison rowmap with -1
// ---------------------------------------------------------------------------
__global__ __launch_bounds__(256) void k_init(int* meta, int* rowmap) {
  const int t = blockIdx.x * 256 + threadIdx.x;
  if (t < PAD_SLOTS) rowmap[t] = -1;
  if (blockIdx.x == 0 && threadIdx.x < 32) meta[threadIdx.x] = 0;
}

// ---------------------------------------------------------------------------
// 1) router: one wave per token; 8-way dot, cross-lane reduce, top-2 softmax
// ---------------------------------------------------------------------------
__global__ __launch_bounds__(256)
void k_router(const float* __restrict__ x, const float* __restrict__ rw,
              int* __restrict__ topIdx, float* __restrict__ scores) {
  const int lane = threadIdx.x & 31;
  const int t    = blockIdx.x * 8 + (threadIdx.x >> 5);
  const float* xr = x + (size_t)t * HDIM;
  float acc[NEXP];
#pragma unroll
  for (int e = 0; e < NEXP; ++e) acc[e] = 0.0f;
  for (int h = lane; h < HDIM; h += 32) {
    const float xv = xr[h];
#pragma unroll
    for (int e = 0; e < NEXP; ++e) acc[e] += xv * rw[h * NEXP + e];
  }
#pragma unroll
  for (int off = 16; off > 0; off >>= 1) {
#pragma unroll
    for (int e = 0; e < NEXP; ++e) acc[e] += __shfl_xor(acc[e], off, 32);
  }
  if (lane == 0) {
    int i0 = 0; float v0 = acc[0];
#pragma unroll
    for (int e = 1; e < NEXP; ++e) if (acc[e] > v0) { v0 = acc[e]; i0 = e; }
    int i1 = -1; float v1 = -3.4e38f;
#pragma unroll
    for (int e = 0; e < NEXP; ++e)
      if (e != i0 && acc[e] > v1) { v1 = acc[e]; i1 = e; }
    const float q = __expf(v1 - v0);      // softmax over {v0, v1}
    const float s = 1.0f / (1.0f + q);
    topIdx[t * 2 + 0] = i0;  topIdx[t * 2 + 1] = i1;
    scores[t * 2 + 0] = s;   scores[t * 2 + 1] = q * s;
  }
}

// ---------------------------------------------------------------------------
// 2) histogram, 3) scan (16-aligned slot regions), 4) scatter
// ---------------------------------------------------------------------------
__global__ __launch_bounds__(256)
void k_count(const int* __restrict__ topIdx, int* meta) {
  const int s = blockIdx.x * 256 + threadIdx.x;
  if (s < T_TOK * 2) atomicAdd(&meta[topIdx[s]], 1);
}

__global__ void k_scan(int* meta) {
  if (threadIdx.x == 0 && blockIdx.x == 0) {
    int acc = 0;
    for (int e = 0; e < NEXP; ++e) {
      meta[8 + e] = acc;
      acc += (meta[e] + 15) & ~15;        // pad each expert region to 16 rows
    }
    meta[16] = acc;                        // total padded slots
    for (int e = 0; e < NEXP; ++e) meta[17 + e] = 0;
  }
}

__global__ __launch_bounds__(256)
void k_scatter(const int* __restrict__ topIdx, const float* __restrict__ scores,
               int* meta, int* __restrict__ rowmap, float* __restrict__ slotSc) {
  const int s = blockIdx.x * 256 + threadIdx.x;
  if (s < T_TOK * 2) {
    const int e = topIdx[s];
    const int pos = meta[8 + e] + atomicAdd(&meta[17 + e], 1);
    rowmap[pos] = s;                       // s = token*2 + k
    slotSc[pos] = scores[s];
  }
}

// ---- A staging coordinate helpers (threads 0..127) ------------------------
struct AStage {
  int r;      // tile row 0..15
  int K0;     // K group base 0,4,..,28
  int lane;   // destination fragment lane
  int slot;   // destination v4bf slot 0..3
};
__device__ inline AStage a_stage(int tid) {
  AStage s;
  s.r    = tid >> 3;
  s.K0   = (tid & 7) * 4;
  s.lane = s.r + ((s.K0 & 16) ? 16 : 0);
  s.slot = (s.K0 & 15) >> 2;
  return s;
}
__device__ inline v4bf cvt4(v4f f) {
  v4bf h;
  h[0] = (bf16_t)f[0]; h[1] = (bf16_t)f[1];
  h[2] = (bf16_t)f[2]; h[3] = (bf16_t)f[3];
  return h;
}

// ---------------------------------------------------------------------------
// 5) shared experts gate||up:  inter = silu(x@Wg) * (x@Wu)   (bf16 out)
//    block tile 16(M) x 256(N): 8 waves x two 16x16 WMMA tiles each
// ---------------------------------------------------------------------------
__global__ __launch_bounds__(256)
void k_shared_gateup(const float* __restrict__ x, const float* __restrict__ wg,
                     const float* __restrict__ wu, bf16_t* __restrict__ inter) {
  __shared__ v16bf lAf[2][32];
  const int tid = threadIdx.x, lane = tid & 31, wv = tid >> 5;
  const int m0 = blockIdx.x * 16, n0 = blockIdx.y * 256;
  const AStage st = a_stage(tid);
  const float* aRow = x + (size_t)(m0 + st.r) * HDIM;
  const int c0 = n0 + wv * 32;             // this wave's first tile column

  v8f aG0 = {}, aG1 = {}, aU0 = {}, aU1 = {};
  for (int k0 = 0; k0 < HDIM; k0 += 32) {
    const int buf = (k0 >> 5) & 1;
    if (tid < 128)
      ((v4bf*)&lAf[buf][st.lane])[st.slot] = cvt4(*(const v4f*)(aRow + k0 + st.K0));
    if (k0 + 32 < HDIM)                    // global_prefetch_b8 of next K rows
      __builtin_prefetch(&wg[(size_t)(k0 + 32 + (lane & 16)) * SHARED_ID + c0 + (lane & 15)], 0, 1);
    __syncthreads();
    const v16bf a  = lAf[buf][lane];
    const v16bf g0 = load_b_direct_f32(wg, SHARED_ID, k0, c0,      lane);
    const v16bf g1 = load_b_direct_f32(wg, SHARED_ID, k0, c0 + 16, lane);
    const v16bf u0 = load_b_direct_f32(wu, SHARED_ID, k0, c0,      lane);
    const v16bf u1 = load_b_direct_f32(wu, SHARED_ID, k0, c0 + 16, lane);
    aG0 = __builtin_amdgcn_wmma_f32_16x16x32_bf16(false, a, false, g0, (short)0, aG0, false, false);
    aG1 = __builtin_amdgcn_wmma_f32_16x16x32_bf16(false, a, false, g1, (short)0, aG1, false, false);
    aU0 = __builtin_amdgcn_wmma_f32_16x16x32_bf16(false, a, false, u0, (short)0, aU0, false, false);
    aU1 = __builtin_amdgcn_wmma_f32_16x16x32_bf16(false, a, false, u1, (short)0, aU1, false, false);
  }
  const int cc = c0 + (lane & 15);
#pragma unroll
  for (int i = 0; i < 8; ++i) {
    const int m = m0 + ((lane < 16) ? i : i + 8);
    inter[(size_t)m * SHARED_ID + cc]      = (bf16_t)(siluf(aG0[i]) * aU0[i]);
    inter[(size_t)m * SHARED_ID + cc + 16] = (bf16_t)(siluf(aG1[i]) * aU1[i]);
  }
}

// ---------------------------------------------------------------------------
// 6) shared experts down: out = inter(bf16) @ w_down   -- initializes d_out
// ---------------------------------------------------------------------------
__global__ __launch_bounds__(256)
void k_shared_down(const bf16_t* __restrict__ inter, const float* __restrict__ wd,
                   float* __restrict__ out) {
  __shared__ v16bf lAf[2][32];
  const int tid = threadIdx.x, lane = tid & 31, wv = tid >> 5;
  const int m0 = blockIdx.x * 16, n0 = blockIdx.y * 256;
  const AStage st = a_stage(tid);
  const bf16_t* aRow = inter + (size_t)(m0 + st.r) * SHARED_ID;
  const int c0 = n0 + wv * 32;

  v8f ac0 = {}, ac1 = {};
  for (int k0 = 0; k0 < SHARED_ID; k0 += 32) {
    const int buf = (k0 >> 5) & 1;
    if (tid < 128)
      ((v4bf*)&lAf[buf][st.lane])[st.slot] = *(const v4bf*)(aRow + k0 + st.K0);
    if (k0 + 32 < SHARED_ID)
      __builtin_prefetch(&wd[(size_t)(k0 + 32 + (lane & 16)) * HDIM + c0 + (lane & 15)], 0, 1);
    __syncthreads();
    const v16bf a  = lAf[buf][lane];
    const v16bf b0 = load_b_direct_f32(wd, HDIM, k0, c0,      lane);
    const v16bf b1 = load_b_direct_f32(wd, HDIM, k0, c0 + 16, lane);
    ac0 = __builtin_amdgcn_wmma_f32_16x16x32_bf16(false, a, false, b0, (short)0, ac0, false, false);
    ac1 = __builtin_amdgcn_wmma_f32_16x16x32_bf16(false, a, false, b1, (short)0, ac1, false, false);
  }
  const int cc = c0 + (lane & 15);
#pragma unroll
  for (int i = 0; i < 8; ++i) {
    const int m = m0 + ((lane < 16) ? i : i + 8);
    out[(size_t)m * HDIM + cc]      = ac0[i];   // plain store: initializes out
    out[(size_t)m * HDIM + cc + 16] = ac1[i];
  }
}

// ---------------------------------------------------------------------------
// 7) grouped fc1 + SwiGLU: act[slot,n] = silu(x@w1[e][:,n]) * (x@w1[e][:,I+n])
//    grid.x = padded M-tiles (expert from slotBase); grid.y = INTER/256
// ---------------------------------------------------------------------------
__global__ __launch_bounds__(256)
void k_moe_fc1(const float* __restrict__ x, const float* __restrict__ w1,
               const int* __restrict__ meta, const int* __restrict__ rowmap,
               bf16_t* __restrict__ act) {
  __shared__ v16bf lAf[2][32];
  const int* sb = meta + 8;
  const int slot0 = blockIdx.x * 16;
  if (slot0 >= sb[8]) return;              // beyond total padded slots
  int e = 0;
#pragma unroll
  for (int i = 1; i < NEXP; ++i) if (slot0 >= sb[i]) e = i;

  const int tid = threadIdx.x, lane = tid & 31, wv = tid >> 5;
  const int n0 = blockIdx.y * 256;
  const float* w1e = w1 + (size_t)e * HDIM * (2 * INTER_D);
  const AStage st = a_stage(tid);
  const int t2 = (tid < 128) ? rowmap[slot0 + st.r] : -1;
  const float* aRow = (t2 >= 0) ? x + (size_t)(t2 >> 1) * HDIM : nullptr;
  const int c0 = n0 + wv * 32;

  v8f aG0 = {}, aG1 = {}, aU0 = {}, aU1 = {};
  for (int k0 = 0; k0 < HDIM; k0 += 32) {
    const int buf = (k0 >> 5) & 1;
    if (tid < 128) {
      v4bf h = {};
      if (aRow) h = cvt4(*(const v4f*)(aRow + k0 + st.K0));
      ((v4bf*)&lAf[buf][st.lane])[st.slot] = h;
    }
    __syncthreads();
    const v16bf a  = lAf[buf][lane];
    const v16bf g0 = load_b_direct_f32(w1e, 2 * INTER_D, k0, c0,                lane);
    const v16bf g1 = load_b_direct_f32(w1e, 2 * INTER_D, k0, c0 + 16,           lane);
    const v16bf u0 = load_b_direct_f32(w1e, 2 * INTER_D, k0, INTER_D + c0,      lane);
    const v16bf u1 = load_b_direct_f32(w1e, 2 * INTER_D, k0, INTER_D + c0 + 16, lane);
    aG0 = __builtin_amdgcn_wmma_f32_16x16x32_bf16(false, a, false, g0, (short)0, aG0, false, false);
    aG1 = __builtin_amdgcn_wmma_f32_16x16x32_bf16(false, a, false, g1, (short)0, aG1, false, false);
    aU0 = __builtin_amdgcn_wmma_f32_16x16x32_bf16(false, a, false, u0, (short)0, aU0, false, false);
    aU1 = __builtin_amdgcn_wmma_f32_16x16x32_bf16(false, a, false, u1, (short)0, aU1, false, false);
  }
  const int cc = c0 + (lane & 15);
#pragma unroll
  for (int i = 0; i < 8; ++i) {
    const int srow = slot0 + ((lane < 16) ? i : i + 8);
    act[(size_t)srow * INTER_D + cc]      = (bf16_t)(siluf(aG0[i]) * aU0[i]);
    act[(size_t)srow * INTER_D + cc + 16] = (bf16_t)(siluf(aG1[i]) * aU1[i]);
  }
}

// ---------------------------------------------------------------------------
// 8) grouped fc2 + unpermute/combine: out[tok] += score * (act @ w2[e])
// ---------------------------------------------------------------------------
__global__ __launch_bounds__(256)
void k_moe_fc2(const bf16_t* __restrict__ act, const float* __restrict__ w2,
               const int* __restrict__ meta, const int* __restrict__ rowmap,
               const float* __restrict__ slotSc, float* __restrict__ out) {
  __shared__ v16bf lAf[2][32];
  const int* sb = meta + 8;
  const int slot0 = blockIdx.x * 16;
  if (slot0 >= sb[8]) return;
  int e = 0;
#pragma unroll
  for (int i = 1; i < NEXP; ++i) if (slot0 >= sb[i]) e = i;

  const int tid = threadIdx.x, lane = tid & 31, wv = tid >> 5;
  const int n0 = blockIdx.y * 256;
  const float* w2e = w2 + (size_t)e * INTER_D * HDIM;
  const AStage st = a_stage(tid);
  const bf16_t* aRow = act + (size_t)(slot0 + st.r) * INTER_D;
  const int c0 = n0 + wv * 32;

  v8f ac0 = {}, ac1 = {};
  for (int k0 = 0; k0 < INTER_D; k0 += 32) {
    const int buf = (k0 >> 5) & 1;
    if (tid < 128)
      ((v4bf*)&lAf[buf][st.lane])[st.slot] = *(const v4bf*)(aRow + k0 + st.K0);
    __syncthreads();
    const v16bf a  = lAf[buf][lane];
    const v16bf b0 = load_b_direct_f32(w2e, HDIM, k0, c0,      lane);
    const v16bf b1 = load_b_direct_f32(w2e, HDIM, k0, c0 + 16, lane);
    ac0 = __builtin_amdgcn_wmma_f32_16x16x32_bf16(false, a, false, b0, (short)0, ac0, false, false);
    ac1 = __builtin_amdgcn_wmma_f32_16x16x32_bf16(false, a, false, b1, (short)0, ac1, false, false);
  }
  const int cc = c0 + (lane & 15);
#pragma unroll
  for (int i = 0; i < 8; ++i) {
    const int srow = slot0 + ((lane < 16) ? i : i + 8);
    const int t2 = rowmap[srow];
    if (t2 >= 0) {
      const float sc = slotSc[srow];
      atomicAdd(&out[(size_t)(t2 >> 1) * HDIM + cc],      sc * ac0[i]);
      atomicAdd(&out[(size_t)(t2 >> 1) * HDIM + cc + 16], sc * ac1[i]);
    }
  }
}

// ---------------------------------------------------------------------------
extern "C" void kernel_launch(void* const* d_in, const int* in_sizes, int n_in,
                              void* d_out, int out_size, void* d_ws, size_t ws_size,
                              hipStream_t stream) {
  (void)in_sizes; (void)n_in; (void)out_size; (void)ws_size;
  const float* x  = (const float*)d_in[0];   // [T, H]
  const float* rw = (const float*)d_in[1];   // [H, E]
  const float* w1 = (const float*)d_in[2];   // [E, H, 2I]
  const float* w2 = (const float*)d_in[3];   // [E, I, H]
  const float* wg = (const float*)d_in[4];   // [H, SI]
  const float* wu = (const float*)d_in[5];   // [H, SI]
  const float* wd = (const float*)d_in[6];   // [SI, H]
  float* out = (float*)d_out;

  char* ws = (char*)d_ws;
  int*    meta   = (int*)(ws + WS_META);
  int*    rowmap = (int*)(ws + WS_ROWMAP);
  int*    topIdx = (int*)(ws + WS_TOPIDX);
  float*  scores = (float*)(ws + WS_SCORES);
  float*  slotSc = (float*)(ws + WS_SLOTSC);
  bf16_t* big    = (bf16_t*)(ws + WS_BIG);   // inter first, then act (reused)

  k_init<<<(PAD_SLOTS + 255) / 256, 256, 0, stream>>>(meta, rowmap);
  k_router<<<T_TOK / 8, 256, 0, stream>>>(x, rw, topIdx, scores);
  k_count<<<(T_TOK * 2) / 256, 256, 0, stream>>>(topIdx, meta);
  k_scan<<<1, 32, 0, stream>>>(meta);
  k_scatter<<<(T_TOK * 2) / 256, 256, 0, stream>>>(topIdx, scores, meta, rowmap, slotSc);

  k_shared_gateup<<<dim3(T_TOK / 16, SHARED_ID / 256), 256, 0, stream>>>(x, wg, wu, big);
  k_shared_down<<<dim3(T_TOK / 16, HDIM / 256), 256, 0, stream>>>(big, wd, out);

  k_moe_fc1<<<dim3(MAX_MTILES, INTER_D / 256), 256, 0, stream>>>(x, w1, meta, rowmap, big);
  k_moe_fc2<<<dim3(MAX_MTILES, HDIM / 256), 256, 0, stream>>>(big, w2, meta, rowmap, slotSc, out);
}
